// SGC_22230750724357
// MI455X (gfx1250) — matrix operand order repaired and checked by reading, not compiled, not run on hardware
//
#include <hip/hip_runtime.h>
#include <hip/hip_bf16.h>
#include <stdint.h>

// ---------------------------------------------------------------------------
// Problem constants (match reference)
// ---------------------------------------------------------------------------
#define NNODE 8192
#define INDIM 512
#define EMBD  256
#define KSLAB 32
#define BROW  264   // LDS row stride in bf16: 528B = 132 dwords == 4 (mod 64 banks)

typedef __attribute__((ext_vector_type(16))) __bf16 v16bf;
typedef __attribute__((ext_vector_type(8)))  __bf16 v8bf;
typedef __attribute__((ext_vector_type(4)))  __bf16 v4bf;
typedef __attribute__((ext_vector_type(8)))  float  v8f;
typedef __attribute__((ext_vector_type(4)))  float  v4f;
typedef __attribute__((ext_vector_type(4)))  int    v4i;
typedef __attribute__((ext_vector_type(4)))  unsigned int v4u;

#define ASG __attribute__((address_space(1)))
#define ASL __attribute__((address_space(3)))

#if defined(__has_builtin)
#  if __has_builtin(__builtin_amdgcn_global_load_async_to_lds_b128)
#    define HAVE_ASYNC_LDS 1
#  endif
#endif

// ---------------------------------------------------------------------------
// 16-byte global -> LDS copy. Async DMA path on CDNA5 (ASYNCcnt-tracked),
// plain load+ds_store fallback otherwise.
// Builtin signature (from clang diagnostic): (int4 AS1*, int4 AS3*, imm, cpol)
// ---------------------------------------------------------------------------
__device__ __forceinline__ void cp16_g2l(const void* g, void* l) {
#ifdef HAVE_ASYNC_LDS
  __builtin_amdgcn_global_load_async_to_lds_b128(
      (ASG v4i*)(uintptr_t)g,
      (ASL v4i*)(uint32_t)(uintptr_t)l,   // low 32 bits of generic = LDS offset
      /*imm offset=*/0, /*cpol=*/0);
#else
  *(v4u*)l = *(const v4u*)g;
#endif
}

__device__ __forceinline__ void async_wait_all() {
#ifdef HAVE_ASYNC_LDS
#  if __has_builtin(__builtin_amdgcn_s_wait_asynccnt)
  __builtin_amdgcn_s_wait_asynccnt(0);
#  else
  asm volatile("s_wait_asynccnt 0x0" ::: "memory");
#  endif
#endif
}

// ---------------------------------------------------------------------------
// fp32 -> bf16 (round to nearest even)
// ---------------------------------------------------------------------------
__device__ __forceinline__ __bf16 f2bf(float f) {
  union { float f; uint32_t u; } v; v.f = f;
  uint32_t u = v.u;
  uint32_t r = (u + 0x7FFFu + ((u >> 16) & 1u)) >> 16;
  unsigned short s = (unsigned short)r;
  return __builtin_bit_cast(__bf16, s);
}

__device__ __forceinline__ v8bf cvt8(const float* p) {
  v4f a = *(const v4f*)p;
  v4f b = *(const v4f*)(p + 4);
  v8bf r;
  r[0] = f2bf(a.x); r[1] = f2bf(a.y); r[2] = f2bf(a.z); r[3] = f2bf(a.w);
  r[4] = f2bf(b.x); r[5] = f2bf(b.y); r[6] = f2bf(b.z); r[7] = f2bf(b.w);
  return r;
}

__device__ __forceinline__ v16bf cat16(v8bf lo, v8bf hi) {
  return __builtin_shufflevector(lo, hi, 0,1,2,3,4,5,6,7,8,9,10,11,12,13,14,15);
}

// ---------------------------------------------------------------------------
// WMMA A-fragment loaders (bf16 16x16x32, per CDNA5 ISA layout)
// A 16x32 (MxK): lanes 0-15 row M=lane, K = kb+{0..7, 16..23}
//                lanes 16-31 row M=lane-16, K = kb+{8..15, 24..31}
// ---------------------------------------------------------------------------
__device__ __forceinline__ v16bf load_a_bf16(const __bf16* A, int ldk, int kb) {
  const int lane = threadIdx.x & 31;
  const int row  = lane & 15;
  const int koff = kb + ((lane >> 4) ? 8 : 0);
  const __bf16* p = A + (size_t)row * ldk + koff;
  v8bf lo = *(const v8bf*)p;          // K koff .. koff+7
  v8bf hi = *(const v8bf*)(p + 16);   // K koff+16 .. koff+23
  return cat16(lo, hi);
}

__device__ __forceinline__ v16bf load_a_f32(const float* A, int ldk, int kb) {
  const int lane = threadIdx.x & 31;
  const int row  = lane & 15;
  const int koff = kb + ((lane >> 4) ? 8 : 0);
  const float* p = A + (size_t)row * ldk + koff;
  v8bf lo = cvt8(p);
  v8bf hi = cvt8(p + 16);
  return cat16(lo, hi);
}

// ---------------------------------------------------------------------------
// GEMM: C[M x 256] = relu(A[M x K] * B[K x 256] + bias)
// One wave per 16-row strip; all 8 waves of a block share a double-buffered
// LDS copy of the 32x256 B k-slab (async global->LDS, padded rows).
// Grid must cover exactly M/16 strips (M % 128 == 0).
// ---------------------------------------------------------------------------
template <bool AF32>
__global__ __launch_bounds__(256)
void gemm_relu_kernel(const void* A_, const __bf16* __restrict__ B,
                      const float* __restrict__ bias,
                      float* __restrict__ C, int M, int K) {
  __shared__ __bf16 Bs[2][KSLAB * BROW];   // 2 x 16.5 KB

  const int wave  = threadIdx.x >> 5;
  const int lane  = threadIdx.x & 31;
  const int strip = blockIdx.x * 8 + wave;

  const __bf16* Abf  = (const __bf16*)A_;
  const float*  Af   = (const float*)A_;
  const size_t  arow = (size_t)strip * 16 * (size_t)K;

  // Cooperative slab copy: B[kb..kb+31][0..255] -> Bs[buf] (1024 x 16B)
  auto copy_slab = [&](int kb, int buf) {
    const char* gbase = (const char*)(B + (size_t)kb * EMBD);
    int tid = threadIdx.x;
#pragma unroll
    for (int i = 0; i < 4; ++i) {
      int idx = tid + i * 256;   // 0..1023
      int row = idx >> 5;        // 32 chunks of 16B per 256-bf16 row
      int ch  = idx & 31;
      cp16_g2l(gbase + (size_t)row * (EMBD * 2) + ch * 16,
               (char*)&Bs[buf][0] + row * (BROW * 2) + ch * 16);
    }
  };

  // B-fragment from LDS: lane -> N column t*16+(lane&15), K = {0..7,16..23} (+8 for lanes 16-31)
  auto load_b_lds = [&](int buf, int t) -> v16bf {
    const int n  = t * 16 + (lane & 15);
    const int kh = (lane >> 4) * 8;
    v16bf b;
#pragma unroll
    for (int s = 0; s < 8; ++s) {
      b[s]     = Bs[buf][(kh + s) * BROW + n];
      b[s + 8] = Bs[buf][(16 + kh + s) * BROW + n];
    }
    return b;
  };

  v8f acc[16] = {};

  copy_slab(0, 0);
  async_wait_all();
  __syncthreads();

  v16bf a_cur = AF32 ? load_a_f32(Af + arow, K, 0) : load_a_bf16(Abf + arow, K, 0);

  const int NK = K / KSLAB;
  for (int kbi = 0; kbi < NK; ++kbi) {
    const int buf = kbi & 1;

    // Software pipeline: issue next slab copy + next A fragment first,
    // then the 16-WMMA burst overlaps both.
    if (kbi + 1 < NK) copy_slab((kbi + 1) * KSLAB, buf ^ 1);

    v16bf a_next = a_cur;
    if (kbi + 1 < NK)
      a_next = AF32 ? load_a_f32(Af + arow, K, (kbi + 1) * KSLAB)
                    : load_a_bf16(Abf + arow, K, (kbi + 1) * KSLAB);

#pragma unroll
    for (int t = 0; t < 16; ++t) {
      v16bf b = load_b_lds(buf, t);
      acc[t] = __builtin_amdgcn_wmma_f32_16x16x32_bf16(
          /*neg_a=*/false, a_cur, /*neg_b=*/false, b,
          /*c_mod=*/(short)0, acc[t], /*reuse_a=*/false, /*reuse_b=*/false);
    }

    a_cur = a_next;
    async_wait_all();   // next slab landed in LDS (this wave's DMA)
    __syncthreads();    // all waves done reading old slab / see new slab
  }

  // Epilogue. C/D layout: VGPR r, lane<16 -> (M=r, N=lane); lane>=16 -> (M=r+8, N=lane-16)
  const int ncol  = lane & 15;
  const int rbase = (lane >> 4) * 8;
#pragma unroll
  for (int t = 0; t < 16; ++t) {
    const int col = t * 16 + ncol;
    const float bv = bias[col];
#pragma unroll
    for (int r = 0; r < 8; ++r) {
      const int row = strip * 16 + rbase + r;
      float v = acc[t][r] + bv;
      C[(size_t)row * EMBD + col] = v > 0.f ? v : 0.f;
    }
  }
}

// ---------------------------------------------------------------------------
// fp32 -> bf16 bulk convert (4 elems / thread)
// ---------------------------------------------------------------------------
__global__ void cvt_bf16_kernel(const float* in, __bf16* out, int n) {
  int i = (blockIdx.x * blockDim.x + threadIdx.x) * 4;
  if (i >= n) return;
  v4f v = *(const v4f*)(in + i);
  v4bf r;
  r[0] = f2bf(v.x); r[1] = f2bf(v.y); r[2] = f2bf(v.z); r[3] = f2bf(v.w);
  *(v4bf*)(out + i) = r;
}

// ---------------------------------------------------------------------------
// Edge scatter: out[dst] += h[src] * w   (one wave per edge, 8 floats/lane)
// ---------------------------------------------------------------------------
__global__ void spmm_scatter_kernel(const float* __restrict__ h,
                                    const int* __restrict__ src,
                                    const int* __restrict__ dst,
                                    const float* __restrict__ w,
                                    float* __restrict__ out, int E) {
  long gid = (long)blockIdx.x * blockDim.x + threadIdx.x;
  int e = (int)(gid >> 5);
  if (e >= E) return;
  int c = ((int)gid & 31) * 8;
  int s = src[e];
  int d = dst[e];
  float we = w[e];
  const float* hp = h + (size_t)s * EMBD + c;
  v4f a = *(const v4f*)hp;
  v4f b = *(const v4f*)(hp + 4);
  float* op = out + (size_t)d * EMBD + c;
  atomicAdd(op + 0, a.x * we);
  atomicAdd(op + 1, a.y * we);
  atomicAdd(op + 2, a.z * we);
  atomicAdd(op + 3, a.w * we);
  atomicAdd(op + 4, b.x * we);
  atomicAdd(op + 5, b.y * we);
  atomicAdd(op + 6, b.z * we);
  atomicAdd(op + 7, b.w * we);
}

// ---------------------------------------------------------------------------
// Column means of h, y1, y2 -> means[3*256]. grid=(rowChunks,3), block=256
// ---------------------------------------------------------------------------
__global__ void col_mean_kernel(const float* __restrict__ h,
                                const float* __restrict__ y1,
                                const float* __restrict__ y2,
                                float* __restrict__ means) {
  const float* srcm = (blockIdx.y == 0) ? h : (blockIdx.y == 1 ? y1 : y2);
  const int col = threadIdx.x;                  // 0..255
  const int rows_per = NNODE / gridDim.x;
  const int r0 = blockIdx.x * rows_per;
  float s = 0.f;
  for (int r = 0; r < rows_per; ++r)
    s += srcm[(size_t)(r0 + r) * EMBD + col];
  atomicAdd(&means[blockIdx.y * EMBD + col], s * (1.0f / (float)NNODE));
}

// ---------------------------------------------------------------------------
// Attention scalars. One block, 192 threads (3 heads x 64).
// means: [xm | y1m | y2m]; concat pairs: [xm,xm], [xm,y1m], [xm,y2m]
// ---------------------------------------------------------------------------
__global__ void attn_kernel(const float* __restrict__ means,
                            const float* __restrict__ W11, const float* __restrict__ b11,
                            const float* __restrict__ W12, const float* __restrict__ b12,
                            const float* __restrict__ W13, const float* __restrict__ b13,
                            float* __restrict__ a_out) {
  __shared__ float red[192];
  __shared__ float e[3];
  const int tid  = threadIdx.x;
  const int head = tid / 64;
  const int j    = tid % 64;
  const float* W   = (head == 0) ? W11 : (head == 1) ? W12 : W13;
  const float* xm  = means;
  const float* oth = (head == 0) ? means : (head == 1) ? (means + EMBD) : (means + 2 * EMBD);
  float s = 0.f;
  for (int i = j; i < EMBD; i += 64)
    s += xm[i] * W[i] + oth[i] * W[EMBD + i];
  red[tid] = s;
  __syncthreads();
  if (j == 0) {
    float t = 0.f;
    for (int k = 0; k < 64; ++k) t += red[head * 64 + k];
    float bb = (head == 0) ? b11[0] : (head == 1) ? b12[0] : b13[0];
    float z = t + bb;
    z = (z > 0.f) ? z : 0.01f * z;   // jax leaky_relu default slope
    e[head] = expf(z);
  }
  __syncthreads();
  if (tid == 0) {
    float denom = e[0] + e[1] + e[2];
    a_out[0] = e[0] / denom;
    a_out[1] = e[1] / denom;
    a_out[2] = e[2] / denom;
  }
}

// ---------------------------------------------------------------------------
// out = sigmoid(a11*h + a12*y1 + a13*y2), vectorized x4
// ---------------------------------------------------------------------------
__global__ void combine_kernel(const float* __restrict__ h,
                               const float* __restrict__ y1,
                               const float* __restrict__ y2,
                               const float* __restrict__ a,
                               float* __restrict__ out, int n) {
  int i = (blockIdx.x * blockDim.x + threadIdx.x) * 4;
  if (i >= n) return;
  const float a11 = a[0], a12 = a[1], a13 = a[2];
  v4f hv = *(const v4f*)(h + i);
  v4f v1 = *(const v4f*)(y1 + i);
  v4f v2 = *(const v4f*)(y2 + i);
  v4f r;
#pragma unroll
  for (int c = 0; c < 4; ++c) {
    float z = a11 * hv[c] + a12 * v1[c] + a13 * v2[c];
    r[c] = 1.0f / (1.0f + expf(-z));
  }
  *(v4f*)(out + i) = r;
}

// ---------------------------------------------------------------------------
// Host launcher
// ---------------------------------------------------------------------------
extern "C" void kernel_launch(void* const* d_in, const int* in_sizes, int n_in,
                              void* d_out, int out_size, void* d_ws, size_t ws_size,
                              hipStream_t stream) {
  const float* x     = (const float*)d_in[0];
  const float* adj   = (const float*)d_in[1];
  const int*   src   = (const int*)d_in[2];
  const int*   dst   = (const int*)d_in[3];
  const float* w     = (const float*)d_in[4];
  const float* W_lin = (const float*)d_in[5];
  const float* b_lin = (const float*)d_in[6];
  const float* W_mlp = (const float*)d_in[7];
  const float* b_mlp = (const float*)d_in[8];
  const float* W_str = (const float*)d_in[9];
  const float* b_str = (const float*)d_in[10];
  const float* W11   = (const float*)d_in[11];
  const float* b11   = (const float*)d_in[12];
  const float* W12   = (const float*)d_in[13];
  const float* b12   = (const float*)d_in[14];
  const float* W13   = (const float*)d_in[15];
  const float* b13   = (const float*)d_in[16];
  float* out = (float*)d_out;
  const int E = in_sizes[2];

  // Workspace carve-up (~46.8 MB total)
  char* p = (char*)d_ws;
  __bf16* xbf   = (__bf16*)p; p += (size_t)NNODE * INDIM * 2;   // 8 MB
  __bf16* wlinb = (__bf16*)p; p += (size_t)INDIM * EMBD * 2;    // 256 KB
  __bf16* wmlpb = (__bf16*)p; p += (size_t)INDIM * EMBD * 2;    // 256 KB
  __bf16* wstrb = (__bf16*)p; p += (size_t)NNODE * EMBD * 2;    // 4 MB
  float*  hA    = (float*)p;  p += (size_t)NNODE * EMBD * 4;    // 8 MB
  float*  hB    = (float*)p;  p += (size_t)NNODE * EMBD * 4;    // 8 MB
  float*  y1    = (float*)p;  p += (size_t)NNODE * EMBD * 4;    // 8 MB
  float*  y2    = (float*)p;  p += (size_t)NNODE * EMBD * 4;    // 8 MB
  float*  means = (float*)p;  p += 3 * EMBD * 4;
  float*  avec  = (float*)p;  p += 256;

  // 1) Pre-convert x and weight matrices to bf16
  cvt_bf16_kernel<<<(NNODE * INDIM / 4 + 255) / 256, 256, 0, stream>>>(x, xbf, NNODE * INDIM);
  cvt_bf16_kernel<<<(INDIM * EMBD / 4 + 255) / 256, 256, 0, stream>>>(W_lin, wlinb, INDIM * EMBD);
  cvt_bf16_kernel<<<(INDIM * EMBD / 4 + 255) / 256, 256, 0, stream>>>(W_mlp, wmlpb, INDIM * EMBD);
  cvt_bf16_kernel<<<(NNODE * EMBD / 4 + 255) / 256, 256, 0, stream>>>(W_str, wstrb, NNODE * EMBD);

  // 2) Dense GEMMs via WMMA bf16: 512 strips of 16 rows, 8 waves/block -> 64 blocks
  const int gemm_blocks = (NNODE / 16) / 8;
  gemm_relu_kernel<false><<<gemm_blocks, 256, 0, stream>>>(xbf, wlinb, b_lin, hA, NNODE, INDIM);
  gemm_relu_kernel<false><<<gemm_blocks, 256, 0, stream>>>(xbf, wmlpb, b_mlp, y1, NNODE, INDIM);
  gemm_relu_kernel<true ><<<gemm_blocks, 256, 0, stream>>>(adj, wstrb, b_str, y2, NNODE, NNODE);

  // 3) K=3 scatter rounds (ping-pong hA <-> hB); final result lands in hB
  float* cur = hA;
  float* nxt = hB;
  const long spmm_threads = (long)E * 32;
  const int  spmm_blocks  = (int)((spmm_threads + 255) / 256);
  for (int k = 0; k < 3; ++k) {
    (void)hipMemsetAsync(nxt, 0, (size_t)NNODE * EMBD * 4, stream);
    spmm_scatter_kernel<<<spmm_blocks, 256, 0, stream>>>(cur, src, dst, w, nxt, E);
    float* t = cur; cur = nxt; nxt = t;
  }
  float* hfin = cur;  // after 3 swaps: hB

  // 4) Column means
  (void)hipMemsetAsync(means, 0, 3 * EMBD * 4, stream);
  dim3 mg(32, 3);
  col_mean_kernel<<<mg, EMBD, 0, stream>>>(hfin, y1, y2, means);

  // 5) Attention scalars (softmax over 3 logits)
  attn_kernel<<<1, 192, 0, stream>>>(means, W11, b11, W12, b12, W13, b13, avec);

  // 6) Final sigmoid combine
  const int n = NNODE * EMBD;
  combine_kernel<<<(n / 4 + 255) / 256, 256, 0, stream>>>(hfin, y1, y2, avec, out, n);
}
